// ConsistencyLoss_57475252355317
// MI455X (gfx1250) — matrix operand order repaired and checked by reading, not compiled, not run on hardware
//
#include <hip/hip_runtime.h>
#include <hip/hip_bf16.h>

// ConsistencyLoss on MI455X (gfx1250, wave32, WMMA).
// Dominant cost: 8192x8192x1024 similarity GEMM (~137 GFLOP) -> f16 WMMA with
// f32 accumulation. Normalized f16 embeddings (16MB) fit in the 192MB L2, so
// this is WMMA-throughput bound. Data movement: double-buffered LDS B-panel
// filled with GLOBAL_LOAD_ASYNC_TO_LDS (ASYNCcnt) when available, so the next
// panel streams in while the current one feeds back-to-back WMMAs.

typedef __attribute__((ext_vector_type(16))) _Float16 v16h;
typedef __attribute__((ext_vector_type(8)))  _Float16 v8h;
typedef __attribute__((ext_vector_type(8)))  float    v8f;

#define BM 128
#define BN 128
#define BK 32
#define BS_PAD 40  // halves per row: 80B = 20 dwords stride -> conflict-free

#if defined(__AMDGCN__) && __has_builtin(__builtin_amdgcn_global_load_async_to_lds_b128)
#define USE_ASYNC_LDS 1
#else
#define USE_ASYNC_LDS 0
#endif

#if USE_ASYNC_LDS
// Probe-derived signature: (AS1 v4i* gsrc, AS3 v4i* ldst, imm offset, imm cpol)
typedef int v4i_t __attribute__((vector_size(16)));
typedef __attribute__((address_space(1))) v4i_t gv4i;
typedef __attribute__((address_space(3))) v4i_t lv4i;
__device__ __forceinline__ void async_ld16(const _Float16* g, _Float16* l) {
  __builtin_amdgcn_global_load_async_to_lds_b128(
      (gv4i*)const_cast<_Float16*>(g), (lv4i*)l, 0, 0);
}
__device__ __forceinline__ void wait_async0() {
#if __has_builtin(__builtin_amdgcn_s_wait_asynccnt)
  __builtin_amdgcn_s_wait_asynccnt(0);
#else
  asm volatile("s_wait_asynccnt 0" ::: "memory");
#endif
}
#endif

// ---------------------------------------------------------------------------
// Kernel 1: one wave (32 lanes) per row.
//  - L2-normalize embedding row, store as f16.
//  - softmax stats over E=16 logits; pack K=32 KL operands:
//      A2[i] = [-L_i (16) | 1.0 | zeros]   B2[j] = [P_j (16) | negent_j | zeros]
//    so   dot(A2[i], B2[j]) = negent[j] - L_i . P_j = KL(P_j || P_i).
// ---------------------------------------------------------------------------
__global__ __launch_bounds__(256) void prep_kernel(
    const float* __restrict__ rp, const float* __restrict__ emb,
    _Float16* __restrict__ embh, _Float16* __restrict__ A2,
    _Float16* __restrict__ B2, float* __restrict__ accum, int B, int H) {
  const int lane = threadIdx.x & 31;
  const int wave = threadIdx.x >> 5;
  const int row  = blockIdx.x * 8 + wave;
  if (blockIdx.x == 0 && threadIdx.x == 0) { accum[0] = 0.0f; accum[1] = 0.0f; }
  if (row >= B) return;

  // --- embedding normalization (H assumed multiple of 32) ---
  const float* er = emb + (size_t)row * H;
  _Float16* orow  = embh + (size_t)row * H;
  const int nt = H >> 5;
  float ss = 0.0f;
  for (int t = 0; t < nt; ++t) { float x = er[t * 32 + lane]; ss += x * x; }
#pragma unroll
  for (int o = 16; o >= 1; o >>= 1) ss += __shfl_xor(ss, o, 32);
  const float inv = rsqrtf(ss);
  for (int t = 0; t < nt; ++t)
    orow[t * 32 + lane] = (_Float16)(er[t * 32 + lane] * inv);

  // --- routing softmax stats (E = 16, lanes 0..15 hold logits) ---
  float v = (lane < 16) ? rp[row * 16 + lane] : -__builtin_inff();
  float m = v;
#pragma unroll
  for (int o = 16; o >= 1; o >>= 1) m = fmaxf(m, __shfl_xor(m, o, 32));
  float e = (lane < 16) ? __expf(v - m) : 0.0f;
  float s = e;
#pragma unroll
  for (int o = 16; o >= 1; o >>= 1) s += __shfl_xor(s, o, 32);
  const float L = v - m - __logf(s);
  const float P = e / s;
  float ne = (lane < 16) ? P * L : 0.0f;
#pragma unroll
  for (int o = 16; o >= 1; o >>= 1) ne += __shfl_xor(ne, o, 32);

  _Float16 av, bv;
  if (lane < 16)        { av = (_Float16)(-L);   bv = (_Float16)P;  }
  else if (lane == 16)  { av = (_Float16)1.0f;   bv = (_Float16)ne; }
  else                  { av = (_Float16)0.0f;   bv = (_Float16)0.0f; }
  A2[row * 32 + lane] = av;
  B2[row * 32 + lane] = bv;
}

// ---------------------------------------------------------------------------
// Kernel 2: 128x128 output tile per 256-thread block (8 waves).
// Wave w owns rows [i0+16w, i0+16w+16) x all 128 cols -> 8 sim accumulators.
// Double-buffered LDS B-panel; A fragments loaded from global (row-private).
// VGPR layouts (A/B: contiguous k order chosen here, not the ISA table order; C/D: ISA layout):
//   A (16x32 f16): lane<16 -> K {kc+0..15};           lane>=16 -> {kc+16..31}
//   B (32x16 f16): lane<16 -> K {kc+0..15};           lane>=16 -> {kc+16..31}
//   C/D f32:       VGPR r, lane l -> (M = r + 8*(l>=16), N = l&15)
// ---------------------------------------------------------------------------
__global__ __launch_bounds__(256) void gemm_kl_kernel(
    const _Float16* __restrict__ embh, const _Float16* __restrict__ A2,
    const _Float16* __restrict__ B2, float* __restrict__ accum, int B, int H) {
  __shared__ _Float16 Bs[2][BN][BS_PAD];
  __shared__ float redS[16];

  const int tid  = threadIdx.x;
  const int lane = tid & 31;
  const int wave = tid >> 5;
  const int i0 = blockIdx.y * BM;
  const int j0 = blockIdx.x * BN;
  const int hi = (lane & 16) ? 1 : 0;        // which half of the wave
  const int asel = hi ? 16 : 0;              // A-fragment K sub-offset
  const int bsel = hi ? 16 : 0;              // B-fragment K sub-offset
  const int rowm = i0 + wave * 16 + (lane & 15);
  const _Float16* arow = embh + (size_t)rowm * H;

  // per-thread B-panel copy coordinates: 2 chunks of 16B (8KB total / block)
  const int r0 = tid >> 2;
  const int c0 = (tid & 3) * 8;
  const _Float16* gsrc0 = embh + (size_t)(j0 + r0) * H + c0;
  const _Float16* gsrc1 = embh + (size_t)(j0 + r0 + 64) * H + c0;

  // --- preload panel 0 into buffer 0 ---
#if USE_ASYNC_LDS
  async_ld16(gsrc0, &Bs[0][r0][c0]);
  async_ld16(gsrc1, &Bs[0][r0 + 64][c0]);
  wait_async0();
#else
  *(v8h*)&Bs[0][r0][c0]      = *(const v8h*)gsrc0;
  *(v8h*)&Bs[0][r0 + 64][c0] = *(const v8h*)gsrc1;
#endif
  __syncthreads();

  const v8f vzero = {};
  v8f acc[8];
#pragma unroll
  for (int n = 0; n < 8; ++n) acc[n] = vzero;

  const int nk = H / BK;
  for (int ks = 0; ks < nk; ++ks) {
    const int cur = ks & 1;
    const int kc  = ks * BK;
    const bool has_next = (ks + 1 < nk);

    // kick off next panel into the other buffer while we compute
#if USE_ASYNC_LDS
    if (has_next) {
      async_ld16(gsrc0 + kc + BK, &Bs[cur ^ 1][r0][c0]);
      async_ld16(gsrc1 + kc + BK, &Bs[cur ^ 1][r0 + 64][c0]);
    }
#else
    v8h st0, st1;
    if (has_next) {
      st0 = *(const v8h*)(gsrc0 + kc + BK);
      st1 = *(const v8h*)(gsrc1 + kc + BK);
    }
#endif

    // A fragment for this k-step (row-private, straight from global/L2)
    union { v16h v; v8h h[2]; } ua;
    ua.h[0] = *(const v8h*)(arow + kc + asel);
    ua.h[1] = *(const v8h*)(arow + kc + asel + 8);

    // batch all 8 B fragments into registers, then chain the WMMAs
    v16h bf[8];
#pragma unroll
    for (int n = 0; n < 8; ++n) {
      union { v16h v; v8h h[2]; } ub;
      const _Float16* bcol = &Bs[cur][n * 16 + (lane & 15)][bsel];
      ub.h[0] = *(const v8h*)bcol;
      ub.h[1] = *(const v8h*)(bcol + 8);
      bf[n] = ub.v;
    }
#pragma unroll
    for (int n = 0; n < 8; ++n)
      acc[n] = __builtin_amdgcn_wmma_f32_16x16x32_f16(
          false, ua.v, false, bf[n], (short)0, acc[n], false, false);

#if USE_ASYNC_LDS
    if (has_next) wait_async0();
#else
    if (has_next) {
      *(v8h*)&Bs[cur ^ 1][r0][c0]      = st0;
      *(v8h*)&Bs[cur ^ 1][r0 + 64][c0] = st1;
    }
#endif
    __syncthreads();
  }

  // KL tile (single K=32 WMMA per subtile) + masked accumulation
  union { v16h v; v8h h[2]; } ka;
  const _Float16* a2r = A2 + (size_t)rowm * 32;
  ka.h[0] = *(const v8h*)(a2r + asel);
  ka.h[1] = *(const v8h*)(a2r + asel + 8);

  float lsum = 0.0f, lcnt = 0.0f;
#pragma unroll
  for (int n = 0; n < 8; ++n) {
    const int gj = j0 + n * 16 + (lane & 15);
    union { v16h v; v8h h[2]; } kb;
    const _Float16* b2r = B2 + (size_t)gj * 32 + bsel;
    kb.h[0] = *(const v8h*)(b2r);
    kb.h[1] = *(const v8h*)(b2r + 8);
    v8f kl = __builtin_amdgcn_wmma_f32_16x16x32_f16(
        false, ka.v, false, kb.v, (short)0, vzero, false, false);
#pragma unroll
    for (int r = 0; r < 8; ++r) {
      const int gi = i0 + wave * 16 + r + (hi ? 8 : 0);
      if (acc[n][r] > 0.8f && gi != gj) { lsum += kl[r]; lcnt += 1.0f; }
    }
  }

  // wave reduce -> LDS -> single atomic pair per block
#pragma unroll
  for (int o = 16; o >= 1; o >>= 1) {
    lsum += __shfl_xor(lsum, o, 32);
    lcnt += __shfl_xor(lcnt, o, 32);
  }
  if (lane == 0) { redS[wave] = lsum; redS[8 + wave] = lcnt; }
  __syncthreads();
  if (tid == 0) {
    float bs = 0.0f, bc = 0.0f;
#pragma unroll
    for (int w = 0; w < 8; ++w) { bs += redS[w]; bc += redS[8 + w]; }
    atomicAdd(&accum[0], bs);
    atomicAdd(&accum[1], bc);
  }
}

__global__ void finalize_kernel(const float* __restrict__ accum,
                                float* __restrict__ out) {
  out[0] = (accum[1] > 0.0f) ? (accum[0] / accum[1]) : 0.0f;
}

// ---------------------------------------------------------------------------
extern "C" void kernel_launch(void* const* d_in, const int* in_sizes, int n_in,
                              void* d_out, int out_size, void* d_ws,
                              size_t ws_size, hipStream_t stream) {
  const float* rp  = (const float*)d_in[0];   // routing_probs [B, 16]
  const float* emb = (const float*)d_in[1];   // input_embeddings [B, H]
  const int B = in_sizes[0] / 16;
  const int H = in_sizes[1] / B;

  // workspace layout: [accum sum,cnt | pad to 256B | emb f16 | A2 | B2]
  float* accum   = (float*)d_ws;
  _Float16* embh = (_Float16*)((char*)d_ws + 256);
  _Float16* A2   = embh + (size_t)B * H;
  _Float16* B2   = A2 + (size_t)B * 32;
  float* out     = (float*)d_out;

  const int prep_blocks = (B + 7) / 8;
  prep_kernel<<<prep_blocks, 256, 0, stream>>>(rp, emb, embh, A2, B2, accum, B, H);

  dim3 grid(B / BN, B / BM);
  gemm_kl_kernel<<<grid, 256, 0, stream>>>(embh, A2, B2, accum, B, H);

  finalize_kernel<<<1, 1, 0, stream>>>(accum, out);
}
